// Head_31653908971734
// MI455X (gfx1250) — compile-verified, matrix operand order checked by reading, not run on hardware
//
#include <hip/hip_runtime.h>

// ---------------------------------------------------------------------------
// Problem constants (match reference)
// ---------------------------------------------------------------------------
#define NPTS 8192
#define MDIM 3
#define CCH  64
#define C2   128
#define KNN  16
#define NEG  0.2f
#define EPSV 1e-6f

typedef __attribute__((ext_vector_type(16))) __bf16 v16bf;
typedef __attribute__((ext_vector_type(8)))  float  v8f;
typedef int v4i_gcc __attribute__((vector_size(16)));   // matches builtin prototype

#if __has_builtin(__builtin_amdgcn_global_load_async_to_lds_b128) && \
    __has_builtin(__builtin_amdgcn_s_wait_asynccnt)
#define HAVE_ASYNC_LDS 1
#else
#define HAVE_ASYNC_LDS 0
#endif

// fp32 -> bf16 round-to-nearest-even
static __device__ __forceinline__ unsigned short f2bf(float f) {
    unsigned u = __float_as_uint(f);
    u = u + 0x7FFFu + ((u >> 16) & 1u);
    return (unsigned short)(u >> 16);
}
static __device__ __forceinline__ unsigned pack2bf(float a, float b) {
    return (unsigned)f2bf(a) | ((unsigned)f2bf(b) << 16);
}

// K index of the bf16 pair held in "VGPR slot" s for the given lane half,
// per CDNA5 ISA 16-bit A layout (16x32): slots 0..3 -> K {0,2,4,6}+8*half,
// slots 4..7 -> K {16,18,20,22}+8*half.
static __device__ __forceinline__ int kpair(int s, int half) {
    return ((s & 4) ? 16 : 0) + ((s & 3) << 1) + (half << 3);
}

// ---------------------------------------------------------------------------
// Async (ASYNCcnt-tracked) global->LDS copy of 16B chunks, with fallback.
// Caller must follow with wait_lds_copies() + __syncthreads().
// ---------------------------------------------------------------------------
static __device__ __forceinline__ void copy_to_lds_16(const void* __restrict__ gsrc,
                                                      void* lds, int bytes,
                                                      int tid, int nthr) {
#if HAVE_ASYNC_LDS
    v4i_gcc __attribute__((address_space(1)))* g =
        (v4i_gcc __attribute__((address_space(1)))*)gsrc;
    v4i_gcc __attribute__((address_space(3)))* l =
        (v4i_gcc __attribute__((address_space(3)))*)lds;
    int chunks = bytes >> 4;
    for (int i = tid; i < chunks; i += nthr)
        __builtin_amdgcn_global_load_async_to_lds_b128(g + i, l + i, 0, 0);
#else
    const uint4* g = (const uint4*)gsrc;
    uint4* l = (uint4*)lds;
    for (int i = tid; i < (bytes >> 4); i += nthr) l[i] = g[i];
#endif
}
static __device__ __forceinline__ void wait_lds_copies() {
#if HAVE_ASYNC_LDS
    __builtin_amdgcn_s_wait_asynccnt(0);
#endif
}

// ---------------------------------------------------------------------------
// D = A(LDS bf16 pair-packed, Rpad x K) @ W(LDS bf16 fragment-packed, K x Ncols)
//   -> D(LDS fp32, Rpad x Ncols)
// A is row-major bf16: u32 word w of row r holds channels (2w, 2w+1).
// A fragment = two ds_load_b128 per lane; B fragment = two ds_load_b128.
// ---------------------------------------------------------------------------
static __device__ void lds_matmul_bf16(const unsigned int* __restrict__ Abf,
                                       const unsigned int* __restrict__ Wlds,
                                       float* __restrict__ D,
                                       int Rpad, int K, int Ncols,
                                       int wave, int nwaves, int lane) {
    int Mtiles = Rpad >> 4;
    int Ntiles = Ncols >> 4;
    int Ktiles = K >> 5;
    int Kw = K >> 1;                 // u32 words per A row
    int total = Mtiles * Ntiles;
    int half = lane >> 4;
    int lr = lane & 15;
    for (int t = wave; t < total; t += nwaves) {
        int mt = t / Ntiles, nt = t - mt * Ntiles;
        int r = mt * 16 + lr;
        v8f acc = {};
        for (int kb = 0; kb < Ktiles; ++kb) {
            union { v16bf v; uint4 q[2]; } af, bfr;
            const unsigned int* ap = Abf + r * Kw + kb * 16 + half * 4;
            af.q[0] = *(const uint4*)ap;        // pairs k = kb*32 + {0,2,4,6} + 8h
            af.q[1] = *(const uint4*)(ap + 8);  // pairs k = kb*32 + 16 + {..} + 8h
            const unsigned int* bp = Wlds + ((((kb * Ntiles + nt) << 5) + lane) << 3);
            bfr.q[0] = *(const uint4*)bp;
            bfr.q[1] = *(const uint4*)(bp + 4);
            acc = __builtin_amdgcn_wmma_f32_16x16x32_bf16(
                false, af.v, false, bfr.v, (short)0, acc, false, false);
        }
        int col = nt * 16 + lr;
#pragma unroll
        for (int j = 0; j < 8; ++j)
            D[(mt * 16 + j + half * 8) * Ncols + col] = acc[j];
    }
}

// ---------------------------------------------------------------------------
// K1: brute-force kNN (self included), top-16 ascending d2, stable ties
// ---------------------------------------------------------------------------
__global__ __launch_bounds__(256) void knn_kernel(const float* __restrict__ x,
                                                  int* __restrict__ idx) {
    int i = blockIdx.x * blockDim.x + threadIdx.x;
    if (i >= NPTS) return;
    float xi0 = x[i * 3 + 0], xi1 = x[i * 3 + 1], xi2 = x[i * 3 + 2];
    float bd[KNN]; int bi[KNN];
#pragma unroll
    for (int q = 0; q < KNN; ++q) { bd[q] = 3.4e38f; bi[q] = 0; }
    for (int j = 0; j < NPTS; ++j) {
        float d0 = xi0 - x[j * 3 + 0];
        float d1 = xi1 - x[j * 3 + 1];
        float d2 = xi2 - x[j * 3 + 2];
        float d = d0 * d0 + d1 * d1 + d2 * d2;
        if (d < bd[KNN - 1]) {
            bd[KNN - 1] = d; bi[KNN - 1] = j;
#pragma unroll
            for (int q = KNN - 1; q > 0; --q) {
                if (bd[q] < bd[q - 1]) {
                    float tf = bd[q]; bd[q] = bd[q - 1]; bd[q - 1] = tf;
                    int ti = bi[q]; bi[q] = bi[q - 1]; bi[q - 1] = ti;
                }
            }
        }
    }
#pragma unroll
    for (int q = 0; q < KNN; ++q) idx[i * KNN + q] = bi[q];
}

// ---------------------------------------------------------------------------
// K2: pack one fp32 KxN weight matrix into bf16 B-fragment order:
// pair index o = ((kb*Ntiles + nt)*32 + lane)*8 + s
//   -> k = kb*32 + kpair(s, lane>>4), n = nt*16 + (lane&15)
//   -> u32 = { w[k*N+n], w[(k+1)*N+n] }
// ---------------------------------------------------------------------------
__global__ __launch_bounds__(256) void pack_kernel(const float* __restrict__ src,
                                                   unsigned int* __restrict__ dst,
                                                   int K, int N) {
    int Ntiles = N >> 4;
    int total = (K >> 5) * Ntiles * 32 * 8;
    for (int o = blockIdx.x * blockDim.x + threadIdx.x; o < total;
         o += gridDim.x * blockDim.x) {
        int s = o & 7;
        int lane = (o >> 3) & 31;
        int t = o >> 8;                 // kb*Ntiles + nt
        int kb = t / Ntiles, nt = t - kb * Ntiles;
        int k = kb * 32 + kpair(s, lane >> 4);
        int n = nt * 16 + (lane & 15);
        dst[o] = pack2bf(src[k * N + n], src[(k + 1) * N + n]);
    }
}

__global__ __launch_bounds__(256) void ones_kernel(float* __restrict__ v) {
    int e = blockIdx.x * blockDim.x + threadIdx.x;
    if (e < NPTS * CCH) v[e] = 1.0f;
}

// ---------------------------------------------------------------------------
// K3..K6: one Kronecker level, fully fused per workgroup:
//   async-stage weight panels into LDS | gather neighbors -> outer-product mean
//   (written bf16 pair-packed) -> @W (WMMA) -> bf16 repack -> @U (WMMA)
//   -> VN-LeakyReLU reductions + elementwise -> write v_out
// ---------------------------------------------------------------------------
__global__ __launch_bounds__(256) void level_kernel(const float* __restrict__ vin,
                                                    float* __restrict__ vout,
                                                    const int* __restrict__ idx,
                                                    const float* __restrict__ x,
                                                    const unsigned int* __restrict__ Wg,
                                                    const unsigned int* __restrict__ Ug,
                                                    int Tin, int Tout, int P, int Rpad) {
    extern __shared__ char smem[];
    unsigned int* abf = (unsigned int*)smem;                 // Rpad*32 words (bf16 A)
    float* vf   = (float*)(smem + (size_t)Rpad * CCH * 2);   // Rpad*64 fp32 (v')
    float* df   = vf + Rpad * CCH;                           // Rpad*64 fp32 (d)
    float* sdot = df + Rpad * CCH;                           // P*64
    float* sdn2 = sdot + P * CCH;                            // P*64
    float* xs   = sdn2 + P * CCH;                            // P*16*3
    int*   nbi  = (int*)(xs + P * KNN * 3);                  // P*16
    unsigned int* wlds = (unsigned int*)(nbi + P * KNN);     // 2*2048 words (W,U)

    int tid = threadIdx.x;
    int lane = tid & 31, wave = tid >> 5, nwaves = blockDim.x >> 5;
    int gp0 = blockIdx.x * P;
    int R = P * Tout;
    int Kw = CCH >> 1;                                       // 32 words / row

    // kick off async weight staging (overlaps with gather/outer-product)
    copy_to_lds_16(Wg, wlds, 2048 * 4, tid, blockDim.x);
    copy_to_lds_16(Ug, wlds + 2048, 2048 * 4, tid, blockDim.x);

    // stage neighbor ids + coords
    for (int e = tid; e < P * KNN; e += blockDim.x) {
        int p = e >> 4, k = e & 15;
        int nb = idx[(gp0 + p) * KNN + k];
        nbi[e] = nb;
        xs[e * 3 + 0] = x[nb * 3 + 0];
        xs[e * 3 + 1] = x[nb * 3 + 1];
        xs[e * 3 + 2] = x[nb * 3 + 2];
    }
    __syncthreads();

    // u[p][a*Tin+b][c] = (1/16) sum_k x_nb[a] * vin[nb][b][c], written bf16-packed
    int elems = P * Tin * Kw;                // one u32 (2 channels) per element
    for (int e = tid; e < elems; e += blockDim.x) {
        int p = e / (Tin * Kw);
        int rem = e - p * Tin * Kw;
        int b = rem / Kw, cp = rem - b * Kw;
        int c0 = cp * 2;
        float a00 = 0.f, a01 = 0.f, a10 = 0.f, a11 = 0.f, a20 = 0.f, a21 = 0.f;
        for (int k = 0; k < KNN; ++k) {
            int nb = nbi[p * KNN + k];
            const float* vp = vin + (nb * Tin + b) * CCH + c0;
            float v0 = vp[0], v1 = vp[1];
            float x0 = xs[(p * KNN + k) * 3 + 0];
            float x1 = xs[(p * KNN + k) * 3 + 1];
            float x2 = xs[(p * KNN + k) * 3 + 2];
            a00 += x0 * v0; a01 += x0 * v1;
            a10 += x1 * v0; a11 += x1 * v1;
            a20 += x2 * v0; a21 += x2 * v1;
        }
        const float inv = 1.0f / (float)KNN;
        int rb = p * Tout;
        abf[(rb + 0 * Tin + b) * Kw + cp] = pack2bf(a00 * inv, a01 * inv);
        abf[(rb + 1 * Tin + b) * Kw + cp] = pack2bf(a10 * inv, a11 * inv);
        abf[(rb + 2 * Tin + b) * Kw + cp] = pack2bf(a20 * inv, a21 * inv);
    }
    // zero the M-tile pad rows
    for (int e = R * Kw + tid; e < Rpad * Kw; e += blockDim.x) abf[e] = 0u;
    wait_lds_copies();
    __syncthreads();

    // v' = u @ W
    lds_matmul_bf16(abf, wlds, vf, Rpad, CCH, CCH, wave, nwaves, lane);
    __syncthreads();
    // repack v' as bf16 A tile
    for (int e = tid; e < Rpad * Kw; e += blockDim.x)
        abf[e] = pack2bf(vf[e * 2], vf[e * 2 + 1]);
    __syncthreads();
    // d = v' @ U
    lds_matmul_bf16(abf, wlds + 2048, df, Rpad, CCH, CCH, wave, nwaves, lane);
    __syncthreads();

    // per-(point, channel) reductions over the Tout rows
    for (int e = tid; e < P * CCH; e += blockDim.x) {
        int p = e >> 6, c = e & 63;
        float dt = 0.f, dn = 0.f;
        for (int t = 0; t < Tout; ++t) {
            int r = p * Tout + t;
            float vv = vf[r * CCH + c];
            float dd = df[r * CCH + c];
            dt += vv * dd;
            dn += dd * dd;
        }
        sdot[e] = dt; sdn2[e] = dn;
    }
    __syncthreads();

    // VN-LeakyReLU elementwise + writeout
    for (int e = tid; e < P * Tout * CCH; e += blockDim.x) {
        int p = e / (Tout * CCH);
        int rem = e - p * Tout * CCH;
        int t = rem >> 6, c = rem & 63;
        int r = p * Tout + t;
        float vv = vf[r * CCH + c];
        float dd = df[r * CCH + c];
        float dt = sdot[p * CCH + c], dn = sdn2[p * CCH + c];
        float proj = dt / (dn + EPSV) * dd;
        float res = NEG * vv + (1.0f - NEG) * ((dt >= 0.f) ? vv : vv - proj);
        vout[((gp0 + p) * Tout + t) * CCH + c] = res;
    }
}

// ---------------------------------------------------------------------------
// K7: mean pool over points -> g[81][64]
// ---------------------------------------------------------------------------
__global__ __launch_bounds__(256) void pool_kernel(const float* __restrict__ v4,
                                                   float* __restrict__ g) {
    int e = blockIdx.x * blockDim.x + threadIdx.x;
    if (e >= 81 * CCH) return;
    float s = 0.f;
    for (int p = 0; p < NPTS; ++p) s += v4[(p * 81) * CCH + e];
    g[e] = s * (1.0f / (float)NPTS);
}

// s[gp][t1][t2][t3][t4][c2] with channel concat [v4 | g]
static __device__ __forceinline__ float fetchS(const float* __restrict__ v4,
                                               const float* __restrict__ g,
                                               int gp, int t1, int t2, int t3, int t4,
                                               int c) {
    int t = ((t1 * 3 + t2) * 3 + t3) * 3 + t4;
    return (c < CCH) ? v4[(gp * 81 + t) * CCH + c] : g[t * CCH + (c - CCH)];
}

// ---------------------------------------------------------------------------
// K8: contraction stage fused: 6-pair trace (bf16-packed) -> @cW (WMMA)
//   -> bf16 repack -> @cU (WMMA) -> VN-LeakyReLU -> diagonal trace * w2 -> out
// ---------------------------------------------------------------------------
__global__ __launch_bounds__(256) void contract_kernel(const float* __restrict__ v4,
                                                       const float* __restrict__ g,
                                                       const float* __restrict__ w4,
                                                       const unsigned int* __restrict__ cWg,
                                                       const unsigned int* __restrict__ cUg,
                                                       const float* __restrict__ w2,
                                                       float* __restrict__ out) {
    const int P = 16, T = 9, Rpad = P * T;    // 144 rows (multiple of 16)
    const int Kw = C2 >> 1;                   // 64 words / row
    extern __shared__ char smem[];
    unsigned int* abf = (unsigned int*)smem;                 // 144*64 words
    float* yf   = (float*)(smem + Rpad * C2 * 2);            // 144*128 fp32
    float* df   = yf + Rpad * C2;                            // 144*128 fp32
    float* sdot = df + Rpad * C2;                            // P*128
    float* sdn2 = sdot + P * C2;                             // P*128
    unsigned int* wlds = (unsigned int*)(sdn2 + P * C2);     // 2*8192 words

    int tid = threadIdx.x;
    int lane = tid & 31, wave = tid >> 5, nwaves = blockDim.x >> 5;
    int gp0 = blockIdx.x * P;

    copy_to_lds_16(cWg, wlds, 8192 * 4, tid, blockDim.x);
    copy_to_lds_16(cUg, wlds + 8192, 8192 * 4, tid, blockDim.x);

    // learned combination of all 6 pairwise traces, two channels per thread
    for (int e = tid; e < P * T * Kw; e += blockDim.x) {
        int p = e / (T * Kw);
        int rem = e - p * T * Kw;
        int ab = rem / Kw, cp = rem - ab * Kw;
        int c0 = cp * 2;
        int A = ab / 3, B = ab - A * 3;
        int gp = gp0 + p;
        float acc0 = 0.f, acc1 = 0.f;
        for (int d = 0; d < 3; ++d) {
#pragma unroll
            for (int cc = 0; cc < 2; ++cc) {
                int c = c0 + cc;
                float a = 0.f;
                a += fetchS(v4, g, gp, d, d, A, B, c) * w4[0 * C2 + c];  // (1,2)
                a += fetchS(v4, g, gp, d, A, d, B, c) * w4[1 * C2 + c];  // (1,3)
                a += fetchS(v4, g, gp, d, A, B, d, c) * w4[2 * C2 + c];  // (1,4)
                a += fetchS(v4, g, gp, A, d, d, B, c) * w4[3 * C2 + c];  // (2,3)
                a += fetchS(v4, g, gp, A, d, B, d, c) * w4[4 * C2 + c];  // (2,4)
                a += fetchS(v4, g, gp, A, B, d, d, c) * w4[5 * C2 + c];  // (3,4)
                if (cc == 0) acc0 += a; else acc1 += a;
            }
        }
        abf[(p * T + ab) * Kw + cp] = pack2bf(acc0, acc1);
    }
    wait_lds_copies();
    __syncthreads();

    // y = hh @ cW
    lds_matmul_bf16(abf, wlds, yf, Rpad, C2, C2, wave, nwaves, lane);
    __syncthreads();
    for (int e = tid; e < Rpad * Kw; e += blockDim.x)
        abf[e] = pack2bf(yf[e * 2], yf[e * 2 + 1]);
    __syncthreads();
    // d = y @ cU
    lds_matmul_bf16(abf, wlds + 8192, df, Rpad, C2, C2, wave, nwaves, lane);
    __syncthreads();

    for (int e = tid; e < P * C2; e += blockDim.x) {
        int p = e >> 7, c = e & 127;
        float dt = 0.f, dn = 0.f;
        for (int t = 0; t < T; ++t) {
            float vv = yf[(p * T + t) * C2 + c];
            float dd = df[(p * T + t) * C2 + c];
            dt += vv * dd;
            dn += dd * dd;
        }
        sdot[e] = dt; sdn2[e] = dn;
    }
    __syncthreads();

    // final k=2 contraction: trace of the 3x3 block of VN-ReLU(y) * w2
    for (int e = tid; e < P * C2; e += blockDim.x) {
        int p = e >> 7, c = e & 127;
        float dt = sdot[e], dn = sdn2[e];
        float sum = 0.f;
        for (int d = 0; d < 3; ++d) {
            int r = p * T + d * 4;               // (d,d) -> d*3+d
            float vv = yf[r * C2 + c];
            float dd = df[r * C2 + c];
            float proj = dt / (dn + EPSV) * dd;
            float res = NEG * vv + (1.0f - NEG) * ((dt >= 0.f) ? vv : vv - proj);
            sum += res;
        }
        out[(gp0 + p) * C2 + c] = sum * w2[c];
    }
}

// ---------------------------------------------------------------------------
// Host launcher
// ---------------------------------------------------------------------------
extern "C" void kernel_launch(void* const* d_in, const int* in_sizes, int n_in,
                              void* d_out, int out_size, void* d_ws, size_t ws_size,
                              hipStream_t stream) {
    (void)in_sizes; (void)n_in; (void)out_size; (void)ws_size;
    const float* x  = (const float*)d_in[0];
    const float* kW = (const float*)d_in[1];
    const float* kU = (const float*)d_in[2];
    const float* w4 = (const float*)d_in[3];
    const float* cW = (const float*)d_in[4];
    const float* cU = (const float*)d_in[5];
    const float* w2 = (const float*)d_in[6];
    float* out = (float*)d_out;

    // workspace layout
    char* ws = (char*)d_ws;
    const size_t vbytes = (size_t)NPTS * 81 * CCH * sizeof(float);  // 169.9 MB
    int*   idx = (int*)ws;                                          // 8192*16 ints
    float* vA  = (float*)(ws + 524288);
    float* vB  = (float*)(ws + 524288 + vbytes);
    unsigned int* wpk = (unsigned int*)(ws + 524288 + 2 * vbytes);  // 32768 words
    float* g   = (float*)(ws + 524288 + 2 * vbytes + 131072);
    unsigned int* kWw = wpk;                 // 4 * 2048 words
    unsigned int* kUw = wpk + 8192;          // 4 * 2048 words
    unsigned int* cWw = wpk + 16384;         // 8192 words
    unsigned int* cUw = wpk + 24576;         // 8192 words

    knn_kernel<<<NPTS / 256, 256, 0, stream>>>(x, idx);
    for (int t = 0; t < 4; ++t) {
        pack_kernel<<<8, 256, 0, stream>>>(kW + t * 4096, kWw + t * 2048, CCH, CCH);
        pack_kernel<<<8, 256, 0, stream>>>(kU + t * 4096, kUw + t * 2048, CCH, CCH);
    }
    pack_kernel<<<32, 256, 0, stream>>>(cW, cWw, C2, C2);
    pack_kernel<<<32, 256, 0, stream>>>(cU, cUw, C2, C2);
    ones_kernel<<<(NPTS * CCH) / 256, 256, 0, stream>>>(vA);

    // Kronecker levels (ping-pong vA/vB)
    const int Tin_[4]  = {1, 3, 9, 27};
    const int Tout_[4] = {3, 9, 27, 81};
    const int P_[4]    = {16, 16, 8, 4};
    const int Rpad_[4] = {48, 144, 224, 336};
    const float* vin_[4] = {vA, vB, vA, vB};
    float*       vout_[4] = {vB, vA, vB, vA};
    for (int t = 0; t < 4; ++t) {
        int P = P_[t], Rpad = Rpad_[t];
        size_t sh = (size_t)Rpad * CCH * 2                 // abf
                  + (size_t)Rpad * CCH * 4 * 2             // vf, df
                  + (size_t)P * CCH * 4 * 2                // sdot, sdn2
                  + (size_t)P * KNN * 3 * 4                // xs
                  + (size_t)P * KNN * 4                    // nbi
                  + 16384;                                 // wlds (W + U)
        level_kernel<<<NPTS / P, 256, sh, stream>>>(
            vin_[t], vout_[t], idx, x, kWw + t * 2048, kUw + t * 2048,
            Tin_[t], Tout_[t], P, Rpad);
    }
    // v4 ends in vA
    pool_kernel<<<(81 * CCH + 255) / 256, 256, 0, stream>>>(vA, g);

    size_t shc = (size_t)144 * C2 * 2                      // abf
               + (size_t)144 * C2 * 4 * 2                  // yf, df
               + (size_t)16 * C2 * 4 * 2                   // sdot, sdn2
               + 65536;                                    // wlds (cW + cU)
    contract_kernel<<<NPTS / 16, 256, shc, stream>>>(vA, g, w4, cWw, cUw, w2, out);
}
